// OSA_Block_hybrid_modified_25555055411283
// MI455X (gfx1250) — compile-verified, matrix operand order checked by reading, not run on hardware
//
#include <hip/hip_runtime.h>

// ---------------------------------------------------------------------------
// OSA hybrid block for MI455X (gfx1250, wave32, WMMA).
// All channel-mixing ops run as f16 WMMA GEMMs (f32 accumulate), fused with
// LN / bias / GELU / SE-scale / residual epilogues. Attention uses
// v_wmma_f32_16x16x32_f16 with LDS-staged fragments.
// ---------------------------------------------------------------------------

#define HW 65536                 // 256*256
#define NB 4                     // batch
#define NPIX (NB * HW)           // 262144 pixels

typedef __attribute__((ext_vector_type(16))) _Float16 v16h;
typedef __attribute__((ext_vector_type(8)))  float    v8f;

__device__ __forceinline__ v8f vzero8() {
  v8f z;
#pragma unroll
  for (int i = 0; i < 8; ++i) z[i] = 0.f;
  return z;
}

__device__ __forceinline__ v8f wmma16(v16h a, v16h b, v8f c) {
  // D = A(16x32 f16) * B(32x16 f16) + C(16x16 f32)
  return __builtin_amdgcn_wmma_f32_16x16x32_f16(false, a, false, b,
                                                (short)0, c, false, false);
}

__device__ __forceinline__ float gelu_f(float x) {
  return 0.5f * x * (1.f + erff(x * 0.70710678118654752440f));
}

// Intra-wave LDS producer->consumer fence (CDNA5 split counters).
__device__ __forceinline__ void wave_lds_sync() {
  __builtin_amdgcn_wave_barrier();
  asm volatile("s_wait_dscnt 0" ::: "memory");
  __builtin_amdgcn_wave_barrier();
}

// --- WMMA 16-bit fragment loaders -----------------------------------------
// A-frag (16xK, lane -> M row) and B-frag (Kx16, lane -> N col) share the
// same per-lane element->K mapping (ISA 7.12.2, 16-bit A 16x32 layout).
// p is row-major [row][ld], row = row_base + (lane&15), elements out of
// [0,kwidth) are zero-padded.
__device__ __forceinline__ v16h frag_ld(const _Float16* __restrict__ p,
                                        int ld, int row_base, int kb, int kwidth) {
  const int lane = threadIdx.x & 31;
  const int r = row_base + (lane & 15);
  const int hs = lane >> 4;
  v16h f;
#pragma unroll
  for (int e = 0; e < 16; ++e) {
    const int vg = e >> 1, lo = e & 1;
    const int k = ((vg < 4) ? (vg * 2) : (16 + (vg - 4) * 2)) + lo + hs * 8 + kb;
    f[e] = (k < kwidth) ? p[r * ld + k] : (_Float16)0.f;
  }
  return f;
}

// Transposed B-frag loader: value(k, n) = p[k*ld + col_base + n].
__device__ __forceinline__ v16h frag_ld_t(const _Float16* __restrict__ p,
                                          int ld, int col_base, int kb, int kwidth) {
  const int lane = threadIdx.x & 31;
  const int cidx = col_base + (lane & 15);
  const int hs = lane >> 4;
  v16h f;
#pragma unroll
  for (int e = 0; e < 16; ++e) {
    const int vg = e >> 1, lo = e & 1;
    const int k = ((vg < 4) ? (vg * 2) : (16 + (vg - 4) * 2)) + lo + hs * 8 + kb;
    f[e] = (k < kwidth) ? p[k * ld + cidx] : (_Float16)0.f;
  }
  return f;
}

// ---------------------------------------------------------------------------
// Generic fused 1x1-conv GEMM:  out[b,o,p] = act(sum_c A(in)[b,c,p]*W[o,c]+bias)
// A(in) optionally = LayerNorm2d(in) or concat(in0,in1) or in*scale[b,c].
// ---------------------------------------------------------------------------
struct GemmP {
  const float* in0;
  const float* in1;      // concat upper 64 channels (CIN==128) or nullptr
  const float* w;        // [COUT][CIN]
  const float* bias;     // or nullptr
  const float* lnw;
  const float* lnb;
  float ln_eps;
  int do_ln;             // only valid for CIN==64
  const float* scale;    // [B][CIN] per-channel scale (SE) or nullptr
  const float* resid;    // [B][COUT][HW] or nullptr
  float* out;
  int gelu;
};

template <int CIN, int COUT>
__global__ __launch_bounds__(128) void k_gemm(GemmP p) {
  constexpr int KC  = (CIN + 31) / 32;
  constexpr int LDA = CIN + 8;
  __shared__ _Float16 sW[COUT * CIN];
  __shared__ _Float16 sA[4 * 16 * LDA];

  const int tid = threadIdx.x, lane = tid & 31, wv = tid >> 5;
  for (int i = tid; i < COUT * CIN; i += 128) sW[i] = (_Float16)p.w[i];
  __syncthreads();

  const int g0  = (blockIdx.x * 4 + wv) * 16;   // 16-pixel tile
  const int b   = g0 >> 16;
  const int pix = g0 & (HW - 1);
  _Float16* A = &sA[wv * 16 * LDA];

  if (CIN == 64 && p.do_ln) {
    // fused LayerNorm2d over 64 channels (2 lanes per pixel)
    float regs[32];
    const int px = lane & 15, part = lane >> 4;
    const float* base = p.in0 + (size_t)b * 64 * HW + pix + px;
    float s = 0.f;
#pragma unroll
    for (int j = 0; j < 32; ++j) {
      float v = base[(size_t)(part * 32 + j) * HW];
      regs[j] = v; s += v;
    }
    s += __shfl_xor(s, 16);
    const float mu = s * (1.f / 64.f);
    float vs = 0.f;
#pragma unroll
    for (int j = 0; j < 32; ++j) { float d = regs[j] - mu; vs += d * d; }
    vs += __shfl_xor(vs, 16);
    const float inv = rsqrtf(vs * (1.f / 64.f) + p.ln_eps);
#pragma unroll
    for (int j = 0; j < 32; ++j) {
      const int c = part * 32 + j;
      A[px * LDA + c] = (_Float16)((regs[j] - mu) * inv * p.lnw[c] + p.lnb[c]);
    }
  } else {
    constexpr int C0 = (CIN == 128) ? 64 : CIN;
    for (int i = lane; i < 16 * CIN; i += 32) {
      const int px = i & 15, c = i >> 4;
      const float* src = p.in0;
      int cc = c;
      if (CIN == 128 && c >= 64) { src = p.in1; cc = c - 64; }
      float v = src[((size_t)b * C0 + cc) * HW + pix + px];
      if (p.scale) v *= p.scale[b * CIN + c];
      A[px * LDA + c] = (_Float16)v;
    }
  }
  wave_lds_sync();

  v16h af[KC];
#pragma unroll
  for (int kc = 0; kc < KC; ++kc) af[kc] = frag_ld(A, LDA, 0, kc * 32, CIN);

  const int n = lane & 15, half = lane >> 4;
  float* st = (float*)A;  // reuse A-tile region as 16x17 f32 staging

#pragma unroll
  for (int nt = 0; nt < COUT / 16; ++nt) {
    v8f acc = vzero8();
#pragma unroll
    for (int kc = 0; kc < KC; ++kc)
      acc = wmma16(af[kc], frag_ld(sW, CIN, nt * 16, kc * 32, CIN), acc);

    const float bv = p.bias ? p.bias[nt * 16 + n] : 0.f;
#pragma unroll
    for (int v = 0; v < 8; ++v) {
      const int m = v + 8 * half;
      float val = acc[v] + bv;
      if (p.gelu) val = gelu_f(val);
      st[n * 17 + m] = val;
    }
    wave_lds_sync();
#pragma unroll
    for (int r = 0; r < 8; ++r) {
      const int idx = r * 32 + lane;
      const int m = idx & 15, nn = idx >> 4;
      const size_t o = ((size_t)b * COUT + nt * 16 + nn) * HW + pix + m;
      float val = st[nn * 17 + m];
      if (p.resid) val += p.resid[o];
      p.out[o] = val;
    }
    wave_lds_sync();
  }
}

// ---------------------------------------------------------------------------
// LayerNorm over channel dim (standalone, for reused xf tensor)
// ---------------------------------------------------------------------------
__global__ __launch_bounds__(256) void k_ln2d(const float* __restrict__ in,
                                              const float* __restrict__ w,
                                              const float* __restrict__ bvec,
                                              float eps, float* __restrict__ out) {
  const int g = blockIdx.x * 256 + threadIdx.x;
  const int b = g >> 16, pix = g & (HW - 1);
  const float* base = in + (size_t)b * 64 * HW + pix;
  float r[64]; float s = 0.f;
#pragma unroll 8
  for (int c = 0; c < 64; ++c) { r[c] = base[(size_t)c * HW]; s += r[c]; }
  const float mu = s * (1.f / 64.f);
  float vs = 0.f;
#pragma unroll 8
  for (int c = 0; c < 64; ++c) { float d = r[c] - mu; vs += d * d; }
  const float inv = rsqrtf(vs * (1.f / 64.f) + eps);
  float* ob = out + (size_t)b * 64 * HW + pix;
#pragma unroll 8
  for (int c = 0; c < 64; ++c) ob[(size_t)c * HW] = (r[c] - mu) * inv * w[c] + bvec[c];
}

// ---------------------------------------------------------------------------
// Depthwise 3x3 (zero pad 1), optional bias + GELU
// ---------------------------------------------------------------------------
__global__ __launch_bounds__(256) void k_dw3x3(const float* __restrict__ in,
                                               const float* __restrict__ w,
                                               const float* __restrict__ bias,
                                               float* __restrict__ out,
                                               int C, int act) {
  const size_t idx = (size_t)blockIdx.x * 256 + threadIdx.x;
  const int pix = (int)(idx & (HW - 1));
  const int bc = (int)(idx >> 16);
  const int c = bc % C;
  const int y = pix >> 8, x = pix & 255;
  const float* ip = in + ((size_t)bc << 16);
  const float* wp = w + c * 9;
  float acc = bias ? bias[c] : 0.f;
#pragma unroll
  for (int ky = 0; ky < 3; ++ky) {
    const int iy = y + ky - 1;
    if (iy < 0 || iy > 255) continue;
#pragma unroll
    for (int kx = 0; kx < 3; ++kx) {
      const int ix = x + kx - 1;
      if (ix < 0 || ix > 255) continue;
      acc += wp[ky * 3 + kx] * ip[iy * 256 + ix];
    }
  }
  if (act) acc = gelu_f(acc);
  out[idx] = acc;
}

// Depthwise 3x3 on 128 channels + gated split: out[c] = gelu(t[c]) * t[c+64]
__global__ __launch_bounds__(256) void k_dwgate(const float* __restrict__ in,
                                                const float* __restrict__ w,
                                                float* __restrict__ out) {
  const size_t idx = (size_t)blockIdx.x * 256 + threadIdx.x; // over B*64*HW
  const int pix = (int)(idx & (HW - 1));
  const int bc = (int)(idx >> 16);
  const int c = bc & 63, b = bc >> 6;
  const int y = pix >> 8, x = pix & 255;
  const float* ia = in + ((size_t)b * 128 + c) * HW;
  const float* ig = ia + (size_t)64 * HW;
  const float* wa = w + c * 9;
  const float* wg = w + (c + 64) * 9;
  float a = 0.f, g = 0.f;
#pragma unroll
  for (int ky = 0; ky < 3; ++ky) {
    const int iy = y + ky - 1;
    if (iy < 0 || iy > 255) continue;
#pragma unroll
    for (int kx = 0; kx < 3; ++kx) {
      const int ix = x + kx - 1;
      if (ix < 0 || ix > 255) continue;
      const int o = iy * 256 + ix;
      a += wa[ky * 3 + kx] * ia[o];
      g += wg[ky * 3 + kx] * ig[o];
    }
  }
  out[idx] = gelu_f(a) * g;
}

// ---------------------------------------------------------------------------
// Squeeze-Excite
// ---------------------------------------------------------------------------
__global__ __launch_bounds__(256) void k_se_reduce(const float* __restrict__ t,
                                                   float* __restrict__ spool) {
  __shared__ float red[256];
  const float* p = t + (size_t)blockIdx.x * HW;
  float s = 0.f;
  for (int i = threadIdx.x; i < HW; i += 256) s += p[i];
  red[threadIdx.x] = s;
  __syncthreads();
  for (int o = 128; o > 0; o >>= 1) {
    if (threadIdx.x < o) red[threadIdx.x] += red[threadIdx.x + o];
    __syncthreads();
  }
  if (threadIdx.x == 0) spool[blockIdx.x] = red[0] * (1.f / HW);
}

__global__ __launch_bounds__(64) void k_se_mlp(const float* __restrict__ spool,
                                               const float* __restrict__ w1,
                                               const float* __restrict__ w2,
                                               float* __restrict__ s2) {
  __shared__ float sv[64], h[16];
  const int tid = threadIdx.x, b = blockIdx.x;
  sv[tid] = spool[b * 64 + tid];
  __syncthreads();
  if (tid < 16) {
    float a = 0.f;
    for (int k = 0; k < 64; ++k) a += w1[tid * 64 + k] * sv[k];
    h[tid] = a / (1.f + expf(-a));   // silu
  }
  __syncthreads();
  float a = 0.f;
  for (int k = 0; k < 16; ++k) a += w2[tid * 16 + k] * h[k];
  s2[b * 64 + tid] = 1.f / (1.f + expf(-a));
}

// ---------------------------------------------------------------------------
// Window self-attention (8x8 windows, 4 heads, dim 8 zero-padded to 16).
// One wave per (window, head). qkv: (B,96,HW) planes (q:0..31,k:32..63,v:64..95)
// ---------------------------------------------------------------------------
__global__ __launch_bounds__(128) void k_winattn(const float* __restrict__ qkv,
                                                 float* __restrict__ att) {
  __shared__ _Float16 SQ[4][64 * 16];
  __shared__ _Float16 SK[4][64 * 16];
  __shared__ _Float16 SV[4][16 * 68];
  __shared__ _Float16 SS[4][16 * 72];
  const int lane = threadIdx.x & 31, h = threadIdx.x >> 5;
  const int win = blockIdx.x;           // b*1024 + wx*32 + wy
  const int b = win >> 10, wx = (win >> 5) & 31, wy = win & 31;
  const int pbase = (wx * 8) * 256 + wy * 8;
  _Float16* q  = SQ[h];
  _Float16* kk = SK[h];
  _Float16* vv = SV[h];
  _Float16* ss = SS[h];

  for (int i = lane; i < 64 * 16; i += 32) { q[i] = (_Float16)0.f; kk[i] = (_Float16)0.f; }
  for (int i = lane; i < 16 * 68; i += 32) vv[i] = (_Float16)0.f;
  wave_lds_sync();

  for (int i = lane; i < 512; i += 32) {     // 64 tokens x 8 dims
    const int t = i & 63, d = i >> 6;
    const int pix = pbase + (t >> 3) * 256 + (t & 7);
    const size_t pq = ((size_t)b * 96 + h * 8 + d) * HW + pix;
    q[t * 16 + d]  = (_Float16)(qkv[pq] * 0.25f);          // scale = 16^-0.5
    kk[t * 16 + d] = (_Float16)qkv[pq + (size_t)32 * HW];
    vv[d * 68 + t] = (_Float16)qkv[pq + (size_t)64 * HW];
  }
  wave_lds_sync();

  v16h bK[4], bV0, bV1;
#pragma unroll
  for (int nt = 0; nt < 4; ++nt) bK[nt] = frag_ld(kk, 16, nt * 16, 0, 16);
  bV0 = frag_ld(vv, 68, 0, 0, 64);
  bV1 = frag_ld(vv, 68, 0, 32, 64);

  const int nl = lane & 15, half = lane >> 4;
#pragma unroll
  for (int mt = 0; mt < 4; ++mt) {
    const v16h aQ = frag_ld(q, 16, mt * 16, 0, 16);
    v8f s[4];
#pragma unroll
    for (int nt = 0; nt < 4; ++nt) s[nt] = wmma16(aQ, bK[nt], vzero8());
    // row softmax over the 64 columns
#pragma unroll
    for (int v = 0; v < 8; ++v) {
      float rm = -1e30f;
#pragma unroll
      for (int nt = 0; nt < 4; ++nt) rm = fmaxf(rm, s[nt][v]);
#pragma unroll
      for (int m = 1; m < 16; m <<= 1) rm = fmaxf(rm, __shfl_xor(rm, m));
      float sum = 0.f;
#pragma unroll
      for (int nt = 0; nt < 4; ++nt) { float e = expf(s[nt][v] - rm); s[nt][v] = e; sum += e; }
#pragma unroll
      for (int m = 1; m < 16; m <<= 1) sum += __shfl_xor(sum, m);
      const float inv = 1.f / sum;
      const int mloc = v + 8 * half;
#pragma unroll
      for (int nt = 0; nt < 4; ++nt)
        ss[mloc * 72 + nt * 16 + nl] = (_Float16)(s[nt][v] * inv);
    }
    wave_lds_sync();
    const v16h aS0 = frag_ld(ss, 72, 0, 0, 64);
    const v16h aS1 = frag_ld(ss, 72, 0, 32, 64);
    const v8f o = wmma16(aS1, bV1, wmma16(aS0, bV0, vzero8()));
    if (nl < 8) {
#pragma unroll
      for (int v = 0; v < 8; ++v) {
        const int tok = mt * 16 + v + 8 * half;
        const int pix = pbase + (tok >> 3) * 256 + (tok & 7);
        att[((size_t)b * 32 + h * 8 + nl) * HW + pix] = o[v];
      }
    }
    wave_lds_sync();
  }
}

// ---------------------------------------------------------------------------
// Windowed channel (transposed) attention. One wave per (window, head).
// qkv2: (B,192,HW).  16x16 Gram via WMMA, softmax, O = A*V via WMMA.
// ---------------------------------------------------------------------------
__global__ __launch_bounds__(128) void k_chattn(const float* __restrict__ qkv,
                                                const float* __restrict__ temp,
                                                float* __restrict__ oc) {
  __shared__ _Float16 SQ[4][16 * 68];
  __shared__ _Float16 SK[4][16 * 68];
  __shared__ _Float16 SV[4][16 * 68];
  __shared__ _Float16 SS[4][16 * 36];
  const int lane = threadIdx.x & 31, h = threadIdx.x >> 5;
  const int win = blockIdx.x;
  const int b = win >> 10, wx = (win >> 5) & 31, wy = win & 31;
  const int pbase = (wx * 8) * 256 + wy * 8;
  _Float16* q  = SQ[h];
  _Float16* kk = SK[h];
  _Float16* vv = SV[h];
  _Float16* ss = SS[h];

  for (int i = lane; i < 1024; i += 32) {    // 16 rows x 64 tokens
    const int t = i & 63, d = i >> 6;
    const int pix = pbase + (t >> 3) * 256 + (t & 7);
    const size_t pq = ((size_t)b * 192 + h * 16 + d) * HW + pix;
    q[d * 68 + t]  = (_Float16)qkv[pq];
    kk[d * 68 + t] = (_Float16)qkv[pq + (size_t)64 * HW];
    vv[d * 68 + t] = (_Float16)qkv[pq + (size_t)128 * HW];
  }
  wave_lds_sync();

  // L2-normalize q/k rows; fold per-head temperature into q.
  {
    const int d = lane & 15, part = lane >> 4;
    float sq = 0.f, sk = 0.f;
#pragma unroll
    for (int j = 0; j < 32; ++j) {
      const int t = part * 32 + j;
      const float a = (float)q[d * 68 + t];  sq += a * a;
      const float c = (float)kk[d * 68 + t]; sk += c * c;
    }
    sq += __shfl_xor(sq, 16);
    sk += __shfl_xor(sk, 16);
    const float qs = temp[h] / fmaxf(sqrtf(sq), 1e-12f);
    const float ksc = 1.f / fmaxf(sqrtf(sk), 1e-12f);
#pragma unroll
    for (int j = 0; j < 32; ++j) {
      const int t = part * 32 + j;
      q[d * 68 + t]  = (_Float16)((float)q[d * 68 + t] * qs);
      kk[d * 68 + t] = (_Float16)((float)kk[d * 68 + t] * ksc);
    }
  }
  wave_lds_sync();

  v8f s = wmma16(frag_ld(q, 68, 0, 0, 64), frag_ld(kk, 68, 0, 0, 64), vzero8());
  s = wmma16(frag_ld(q, 68, 0, 32, 64), frag_ld(kk, 68, 0, 32, 64), s);

  const int nl = lane & 15, half = lane >> 4;
#pragma unroll
  for (int v = 0; v < 8; ++v) {
    float rm = s[v];
#pragma unroll
    for (int m = 1; m < 16; m <<= 1) rm = fmaxf(rm, __shfl_xor(rm, m));
    float e = expf(s[v] - rm);
    float sum = e;
#pragma unroll
    for (int m = 1; m < 16; m <<= 1) sum += __shfl_xor(sum, m);
    ss[(v + 8 * half) * 36 + nl] = (_Float16)(e / sum);
  }
  wave_lds_sync();

  const v16h aS = frag_ld(ss, 36, 0, 0, 16);
#pragma unroll
  for (int nt = 0; nt < 4; ++nt) {
    const v8f o = wmma16(aS, frag_ld_t(vv, 68, nt * 16, 0, 16), vzero8());
#pragma unroll
    for (int v = 0; v < 8; ++v) {
      const int d = v + 8 * half;             // output channel within head
      const int tok = nt * 16 + nl;
      const int pix = pbase + (tok >> 3) * 256 + (tok & 7);
      oc[((size_t)b * 64 + h * 16 + d) * HW + pix] = o[v];
    }
  }
}

// ---------------------------------------------------------------------------
// Cross-attention branch helpers (tiny, VALU)
// ---------------------------------------------------------------------------
__global__ __launch_bounds__(256) void k_conv_s8(const float* __restrict__ in,
                                                 const float* __restrict__ w,
                                                 const float* __restrict__ bias,
                                                 float* __restrict__ out) {
  const int idx = blockIdx.x * 256 + threadIdx.x;   // 4*32*1024
  const int ox = idx & 31, oy = (idx >> 5) & 31, c = (idx >> 10) & 31, b = idx >> 15;
  float acc = bias[c];
  for (int ci = 0; ci < 32; ++ci) {
    const float* ip = in + ((size_t)b * 32 + ci) * HW;
    const float* wp = w + (c * 32 + ci) * 9;
#pragma unroll
    for (int ky = 0; ky < 3; ++ky) {
      const int iy = oy * 8 + ky - 1;
      if (iy < 0 || iy > 255) continue;
#pragma unroll
      for (int kx = 0; kx < 3; ++kx) {
        const int ix = ox * 8 + kx - 1;
        if (ix < 0 || ix > 255) continue;
        acc += wp[ky * 3 + kx] * ip[iy * 256 + ix];
      }
    }
  }
  out[idx] = acc;
}

__global__ __launch_bounds__(256) void k_conv_s16(const float* __restrict__ in,
                                                  const float* __restrict__ w,
                                                  const float* __restrict__ bias,
                                                  float* __restrict__ out) {
  const int idx = blockIdx.x * 256 + threadIdx.x;   // 4*64*256
  const int ox = idx & 15, oy = (idx >> 4) & 15, c = (idx >> 8) & 63, b = idx >> 14;
  float acc = bias[c];
  for (int ci = 0; ci < 64; ++ci) {
    const float* ip = in + ((size_t)b * 64 + ci) * HW;
    const float* wp = w + (c * 64 + ci) * 9;
#pragma unroll
    for (int ky = 0; ky < 3; ++ky) {
      const int iy = oy * 16 + ky - 1;
      if (iy < 0 || iy > 255) continue;
#pragma unroll
      for (int kx = 0; kx < 3; ++kx) {
        const int ix = ox * 16 + kx - 1;
        if (ix < 0 || ix > 255) continue;
        acc += wp[ky * 3 + kx] * ip[iy * 256 + ix];
      }
    }
  }
  out[idx] = acc;   // (B,64,16,16) == [b][c][256 tokens]
}

// LN(eps 1e-5) + gelu + kv1 linear over 64 channels, per token.
__global__ __launch_bounds__(64) void k_kv(const float* __restrict__ sr,
                                           const float* __restrict__ lnw,
                                           const float* __restrict__ lnb,
                                           const float* __restrict__ kvw,
                                           const float* __restrict__ kvbias,
                                           float* __restrict__ kvout) {
  __shared__ float xr[64], red[64];
  const int tid = threadIdx.x;
  const int b = blockIdx.x >> 8, t = blockIdx.x & 255;
  const float v = sr[((size_t)b * 64 + tid) * 256 + t];
  red[tid] = v; __syncthreads();
  for (int o = 32; o > 0; o >>= 1) { if (tid < o) red[tid] += red[tid + o]; __syncthreads(); }
  const float mu = red[0] * (1.f / 64.f);
  __syncthreads();
  const float d = v - mu;
  red[tid] = d * d; __syncthreads();
  for (int o = 32; o > 0; o >>= 1) { if (tid < o) red[tid] += red[tid + o]; __syncthreads(); }
  const float var = red[0] * (1.f / 64.f);
  __syncthreads();
  xr[tid] = gelu_f((v - mu) * rsqrtf(var + 1e-5f) * lnw[tid] + lnb[tid]);
  __syncthreads();
  float acc = kvbias[tid];
  for (int k = 0; k < 64; ++k) acc += kvw[tid * 64 + k] * xr[k];
  kvout[((size_t)b * 256 + t) * 64 + tid] = acc;
}

// Low-res cross attention: one wave per (b, head, query).
__global__ __launch_bounds__(32) void k_crossattn(const float* __restrict__ iq,
                                                  const float* __restrict__ kvb,
                                                  float* __restrict__ o1low) {
  __shared__ float qsh[16];
  __shared__ float om[32][16];
  const int lane = threadIdx.x;
  const int id = blockIdx.x;                 // b*2048 + h*1024 + qi
  const int qi = id & 1023, h = (id >> 10) & 1, b = id >> 11;
  if (lane < 16) qsh[lane] = iq[(size_t)b * 32768 + (qi * 2 + h) * 16 + lane];
  wave_lds_sync();
  float sreg[8];
  float mx = -1e30f;
#pragma unroll
  for (int jj = 0; jj < 8; ++jj) {
    const int j = jj * 32 + lane;
    const float* kr = kvb + ((size_t)b * 256 + j) * 64 + h * 16;
    float s = 0.f;
#pragma unroll
    for (int d = 0; d < 16; ++d) s += qsh[d] * kr[d];
    s *= 0.25f;
    sreg[jj] = s;
    mx = fmaxf(mx, s);
  }
#pragma unroll
  for (int m = 1; m < 32; m <<= 1) mx = fmaxf(mx, __shfl_xor(mx, m));
  float sum = 0.f;
#pragma unroll
  for (int jj = 0; jj < 8; ++jj) { sreg[jj] = expf(sreg[jj] - mx); sum += sreg[jj]; }
#pragma unroll
  for (int m = 1; m < 32; m <<= 1) sum += __shfl_xor(sum, m);
  const float inv = 1.f / sum;
  float op[16];
#pragma unroll
  for (int d = 0; d < 16; ++d) op[d] = 0.f;
#pragma unroll
  for (int jj = 0; jj < 8; ++jj) {
    const int j = jj * 32 + lane;
    const float* vr = kvb + ((size_t)b * 256 + j) * 64 + (2 + h) * 16;
    const float pp = sreg[jj] * inv;
#pragma unroll
    for (int d = 0; d < 16; ++d) op[d] += pp * vr[d];
  }
#pragma unroll
  for (int d = 0; d < 16; ++d) om[lane][d] = op[d];
  wave_lds_sync();
  if (lane < 16) {
    float a = 0.f;
    for (int l = 0; l < 32; ++l) a += om[l][lane];
    o1low[((size_t)b * 32 + h * 16 + lane) * 1024 + qi] = a;
  }
}

// x8 nearest upsample fused into cu 3x3 conv (32 -> 32 ch, full res)
__global__ __launch_bounds__(256) void k_cu(const float* __restrict__ low,
                                            const float* __restrict__ w,
                                            const float* __restrict__ bias,
                                            float* __restrict__ out) {
  const size_t idx = (size_t)blockIdx.x * 256 + threadIdx.x;  // 4*32*65536
  const int x = (int)(idx & 255), y = (int)((idx >> 8) & 255);
  const int c = (int)((idx >> 16) & 31), b = (int)(idx >> 21);
  float acc = bias[c];
  for (int ci = 0; ci < 32; ++ci) {
    const float* lp = low + ((size_t)b * 32 + ci) * 1024;
    const float* wp = w + (c * 32 + ci) * 9;
#pragma unroll
    for (int ky = 0; ky < 3; ++ky) {
      const int iy = y + ky - 1;
      if (iy < 0 || iy > 255) continue;
#pragma unroll
      for (int kx = 0; kx < 3; ++kx) {
        const int ix = x + kx - 1;
        if (ix < 0 || ix > 255) continue;
        acc += wp[ky * 3 + kx] * lp[(iy >> 3) * 32 + (ix >> 3)];
      }
    }
  }
  out[idx] = acc;
}

// y += concat(o2 (32ch), o1_full (32ch))
__global__ __launch_bounds__(256) void k_combine(float* __restrict__ y,
                                                 const float* __restrict__ o2,
                                                 const float* __restrict__ o1f) {
  const size_t idx = (size_t)blockIdx.x * 256 + threadIdx.x;
  const int pix = (int)(idx & (HW - 1));
  const int bc = (int)(idx >> 16);
  const int c = bc & 63, b = bc >> 6;
  const float add = (c < 32)
      ? o2[(((size_t)b * 32 + c) << 16) + pix]
      : o1f[(((size_t)b * 32 + (c - 32)) << 16) + pix];
  y[idx] += add;
}

// ---------------------------------------------------------------------------
// Host-side orchestration
// ---------------------------------------------------------------------------
extern "C" void kernel_launch(void* const* d_in, const int* in_sizes, int n_in,
                              void* d_out, int out_size, void* d_ws, size_t ws_size,
                              hipStream_t stream) {
  (void)in_sizes; (void)n_in; (void)out_size; (void)ws_size;
  const float* x        = (const float*)d_in[0];
  const float* norm_w   = (const float*)d_in[1];
  const float* norm_b   = (const float*)d_in[2];
  const float* conv1_w  = (const float*)d_in[3];
  const float* conv1_b  = (const float*)d_in[4];
  const float* conv2_w  = (const float*)d_in[5];
  const float* conv2_b  = (const float*)d_in[6];
  const float* mb_w1    = (const float*)d_in[7];
  const float* mb_b1    = (const float*)d_in[8];
  const float* mb_dw_w  = (const float*)d_in[9];
  const float* mb_dw_b  = (const float*)d_in[10];
  const float* se_w1    = (const float*)d_in[11];
  const float* se_w2    = (const float*)d_in[12];
  const float* mb_w2    = (const float*)d_in[13];
  const float* mb_b2    = (const float*)d_in[14];
  const float* ln_att_w = (const float*)d_in[15];
  const float* ln_att_b = (const float*)d_in[16];
  const float* q1_w     = (const float*)d_in[17];
  const float* q1_b     = (const float*)d_in[18];
  const float* qkv_w    = (const float*)d_in[19];
  const float* to_out_w = (const float*)d_in[20];
  const float* sr_w     = (const float*)d_in[21];
  const float* sr_b     = (const float*)d_in[22];
  const float* ln_sr_w  = (const float*)d_in[23];
  const float* ln_sr_b  = (const float*)d_in[24];
  const float* kv1_w    = (const float*)d_in[25];
  const float* kv1_b    = (const float*)d_in[26];
  const float* q2_w     = (const float*)d_in[27];
  const float* q2_b     = (const float*)d_in[28];
  const float* cd_w     = (const float*)d_in[29];
  const float* cd_b     = (const float*)d_in[30];
  const float* cu_w     = (const float*)d_in[31];
  const float* cu_b     = (const float*)d_in[32];
  const float* cln1_w   = (const float*)d_in[33];
  const float* cln1_b   = (const float*)d_in[34];
  const float* ff1_in_w = (const float*)d_in[35];
  const float* ff1_dw_w = (const float*)d_in[36];
  const float* ff1_out_w= (const float*)d_in[37];
  const float* cln2_w   = (const float*)d_in[38];
  const float* cln2_b   = (const float*)d_in[39];
  const float* ca_temp  = (const float*)d_in[40];
  const float* ca_qkv_w = (const float*)d_in[41];
  const float* ca_dw_w  = (const float*)d_in[42];
  const float* ca_out_w = (const float*)d_in[43];
  const float* cln3_w   = (const float*)d_in[44];
  const float* cln3_b   = (const float*)d_in[45];
  const float* ff2_in_w = (const float*)d_in[46];
  const float* ff2_dw_w = (const float*)d_in[47];
  const float* ff2_out_w= (const float*)d_in[48];
  float* out = (float*)d_out;

  // workspace layout (floats)
  const size_t S64 = (size_t)NB * 64 * HW;    // 16.78M
  const size_t S32 = S64 / 2;
  const size_t S192 = 3 * S64;
  float* ws    = (float*)d_ws;
  float* bufA  = ws;                 // y (running residual stream)
  float* bufB  = bufA + S64;
  float* bufC  = bufB + S64;         // xf / ln scratch
  float* bufE  = bufC + S64;         // 32-ch
  float* bufF  = bufE + S32;         // 32-ch
  float* bufD  = bufF + S32;         // up to 192-ch
  float* bufD2 = bufD + S192;        // 192-ch
  float* spool = bufD2 + S192;       // 256
  float* s2    = spool + 256;        // 256
  float* srout = s2 + 256;           // 65536
  float* kvb   = srout + 65536;      // 65536
  float* iqlow = kvb + 65536;        // 131072
  float* o1low = iqlow + 131072;     // 131072

  const dim3 GB(NPIX / 64), TB(128);   // GEMM launch shape

  GemmP p{};
  auto reset = [&]() {
    p.in0 = nullptr; p.in1 = nullptr; p.w = nullptr; p.bias = nullptr;
    p.lnw = nullptr; p.lnb = nullptr; p.ln_eps = 0.f; p.do_ln = 0;
    p.scale = nullptr; p.resid = nullptr; p.out = nullptr; p.gelu = 0;
  };

  // 1) y = conv1(LN2d(x))                        -> A
  reset(); p.in0 = x; p.w = conv1_w; p.bias = conv1_b;
  p.lnw = norm_w; p.lnb = norm_b; p.ln_eps = 1e-6f; p.do_ln = 1; p.out = bufA;
  k_gemm<64, 64><<<GB, TB, 0, stream>>>(p);

  // 2) t = gelu(mb1(y))                          -> B
  reset(); p.in0 = bufA; p.w = mb_w1; p.bias = mb_b1; p.gelu = 1; p.out = bufB;
  k_gemm<64, 64><<<GB, TB, 0, stream>>>(p);

  // 3) t = gelu(dw3x3(t))                        -> C
  k_dw3x3<<<dim3(NB * 64 * HW / 256), dim3(256), 0, stream>>>(bufB, mb_dw_w, mb_dw_b, bufC, 64, 1);

  // 4-5) SE
  k_se_reduce<<<dim3(256), dim3(256), 0, stream>>>(bufC, spool);
  k_se_mlp<<<dim3(4), dim3(64), 0, stream>>>(spool, se_w1, se_w2, s2);

  // 6) y = mb2(t * s) + y                        -> A
  reset(); p.in0 = bufC; p.w = mb_w2; p.bias = mb_b2; p.scale = s2;
  p.resid = bufA; p.out = bufA;
  k_gemm<64, 64><<<GB, TB, 0, stream>>>(p);

  // 7) xf = LN(y)  (per-pixel over channels)     -> C
  k_ln2d<<<dim3(NPIX / 256), dim3(256), 0, stream>>>(bufA, ln_att_w, ln_att_b, 1e-5f, bufC);

  // 8) q = q1(xf)                                -> E
  reset(); p.in0 = bufC; p.w = q1_w; p.bias = q1_b; p.out = bufE;
  k_gemm<64, 32><<<GB, TB, 0, stream>>>(p);

  // 9) qkv = qkv_w(q)                            -> D (96ch)
  reset(); p.in0 = bufE; p.w = qkv_w; p.out = bufD;
  k_gemm<32, 96><<<GB, TB, 0, stream>>>(p);

  // 10) window attention                         -> F (32ch)
  k_winattn<<<dim3(4096), dim3(128), 0, stream>>>(bufD, bufF);

  // 11) o2 = to_out(att)                         -> E
  reset(); p.in0 = bufF; p.w = to_out_w; p.out = bufE;
  k_gemm<32, 32><<<GB, TB, 0, stream>>>(p);

  // 12) iq = q2(xf)                              -> F
  reset(); p.in0 = bufC; p.w = q2_w; p.bias = q2_b; p.out = bufF;
  k_gemm<64, 32><<<GB, TB, 0, stream>>>(p);

  // 13-16) cross-attention branch (small)
  k_conv_s8<<<dim3(512), dim3(256), 0, stream>>>(bufF, cd_w, cd_b, iqlow);
  k_conv_s16<<<dim3(256), dim3(256), 0, stream>>>(bufC, sr_w, sr_b, srout);
  k_kv<<<dim3(1024), dim3(64), 0, stream>>>(srout, ln_sr_w, ln_sr_b, kv1_w, kv1_b, kvb);
  k_crossattn<<<dim3(8192), dim3(32), 0, stream>>>(iqlow, kvb, o1low);

  // 17) o1_full = cu(upsample8(o1_low))          -> F
  k_cu<<<dim3(NB * 32 * HW / 256), dim3(256), 0, stream>>>(o1low, cu_w, cu_b, bufF);

  // 18) y += concat(o2, o1_full)                 -> A
  k_combine<<<dim3(NB * 64 * HW / 256), dim3(256), 0, stream>>>(bufA, bufE, bufF);

  // 19-21) gated conv FFN #1 (LN fused into ff_in GEMM)
  reset(); p.in0 = bufA; p.w = ff1_in_w;
  p.lnw = cln1_w; p.lnb = cln1_b; p.ln_eps = 1e-6f; p.do_ln = 1; p.out = bufD;
  k_gemm<64, 128><<<GB, TB, 0, stream>>>(p);
  k_dwgate<<<dim3(NB * 64 * HW / 256), dim3(256), 0, stream>>>(bufD, ff1_dw_w, bufB);
  reset(); p.in0 = bufB; p.w = ff1_out_w; p.resid = bufA; p.out = bufA;
  k_gemm<64, 64><<<GB, TB, 0, stream>>>(p);

  // 22-25) windowed channel attention
  reset(); p.in0 = bufA; p.w = ca_qkv_w;
  p.lnw = cln2_w; p.lnb = cln2_b; p.ln_eps = 1e-6f; p.do_ln = 1; p.out = bufD;
  k_gemm<64, 192><<<GB, TB, 0, stream>>>(p);
  k_dw3x3<<<dim3(NB * 192 * HW / 256), dim3(256), 0, stream>>>(bufD, ca_dw_w, nullptr, bufD2, 192, 0);
  k_chattn<<<dim3(4096), dim3(128), 0, stream>>>(bufD2, ca_temp, bufB);
  reset(); p.in0 = bufB; p.w = ca_out_w; p.resid = bufA; p.out = bufA;
  k_gemm<64, 64><<<GB, TB, 0, stream>>>(p);

  // 26-28) gated conv FFN #2
  reset(); p.in0 = bufA; p.w = ff2_in_w;
  p.lnw = cln3_w; p.lnb = cln3_b; p.ln_eps = 1e-6f; p.do_ln = 1; p.out = bufD;
  k_gemm<64, 128><<<GB, TB, 0, stream>>>(p);
  k_dwgate<<<dim3(NB * 64 * HW / 256), dim3(256), 0, stream>>>(bufD, ff2_dw_w, bufB);
  reset(); p.in0 = bufB; p.w = ff2_out_w; p.resid = bufA; p.out = bufA;
  k_gemm<64, 64><<<GB, TB, 0, stream>>>(p);

  // 29) out = conv2(concat(x, y))                -> d_out
  reset(); p.in0 = x; p.in1 = bufA; p.w = conv2_w; p.bias = conv2_b; p.out = out;
  k_gemm<128, 64><<<GB, TB, 0, stream>>>(p);
}
